// LSTM_35897336660320
// MI455X (gfx1250) — matrix operand order repaired and checked by reading, not compiled, not run on hardware
//
#include <hip/hip_runtime.h>
#include <cstddef>
#include <cstdint>

// ---------------------------------------------------------------------------
// Problem constants (from reference): B=256, T=512, D=64, H=512, OUT=1
// ---------------------------------------------------------------------------
#define BB 256
#define TT 512
#define DD 64
#define HH 512

typedef __attribute__((ext_vector_type(16))) _Float16 v16h;
typedef __attribute__((ext_vector_type(8)))  _Float16 v8h_t;
typedef __attribute__((ext_vector_type(8)))  float    v8f;
typedef __attribute__((ext_vector_type(4)))  unsigned int v4u;
typedef __attribute__((ext_vector_type(8)))  int      v8i;
typedef __attribute__((ext_vector_type(4)))  int      v4i;

__device__ __forceinline__ float sigm(float x) { return 1.0f / (1.0f + __expf(-x)); }

// ---------------------------------------------------------------------------
// WMMA operand loaders (wave32; A/B use this file's contiguous k = 16*hi + s order, same for both; C/D per ISA)
//
// A (16x32 f16): lane L: row M = L&15, hi = L>>4.
//   per lane: halves [k0+hi*16, +16)  (two 16B chunks)
// B (32x16 f16): lane L: col N = L&15, hi = L>>4.
//   16 contiguous halves at [k0+hi*16, +16) of W row (n)
// C/D (16x16 f32): lane L: col N = L&15, VGPR v holds row M = v + 8*(L>>4)
// ---------------------------------------------------------------------------
__device__ __forceinline__ v16h load_a_tile(const _Float16* __restrict__ A, int lda,
                                            int m0, int row, int hi, int k0) {
  union { v16h v; v8h_t p[2]; } u;
  const _Float16* pa = A + (size_t)(m0 + row) * (size_t)lda + (size_t)(k0 + hi * 16);
  u.p[0] = *(const v8h_t*)(pa);
  u.p[1] = *(const v8h_t*)(pa + 8);
  return u.v;
}

__device__ __forceinline__ v16h load_b_tile(const _Float16* __restrict__ W, int ldw,
                                            int nrow, int hi, int k0) {
  return *(const v16h*)(W + (size_t)nrow * (size_t)ldw + (size_t)(k0 + hi * 16));
}

// ---------------------------------------------------------------------------
// Fused LSTM step: gates = A1 @ W1^T + A2 @ W2^T + bias ; pointwise update.
//   A1: [256 x K1] f16 (row stride lda1)   W1: [2048 x K1] f16 row-major (global)
//   A2: [256 x 512] f16 (h_prev)           W2: [2048 x 512] f16 row-major
//   W2's per-block panel (4 gates x 16 rows x 512 halfs = 64KB) is staged into
//   LDS with a single 3-D Tensor-Data-Mover descriptor, then read as ds_load.
//   bias: [2048] f32 (b_ih+b_hh)  c: [256x512] f32 (in/out)  hout: [256x512] f16
// One wave handles one 16x16 tile of H for all 4 gates.
// grid = (4, 32), block = 128 (4 waves) -> 16 m-tiles x 32 n-tiles.
// ---------------------------------------------------------------------------
template <int K1>
__global__ __launch_bounds__(128) void lstm_step_kernel(
    const _Float16* __restrict__ A1, int lda1,
    const _Float16* __restrict__ W1,
    const _Float16* __restrict__ A2,
    const _Float16* __restrict__ W2,
    const float* __restrict__ bias,
    float* __restrict__ c,
    _Float16* __restrict__ hout) {
  // W2 panel in LDS: [gate][row(16)][k(512)] halves, 64KB. Only LDS object in
  // this kernel -> static LDS offset 0 (used as D#.lds_addr).
  __shared__ __align__(128) _Float16 sB[4 * 16 * HH];

  const int lane = threadIdx.x & 31;
  const int wave = threadIdx.x >> 5;
  const int m0 = (blockIdx.x * 4 + wave) * 16;
  const int n0 = blockIdx.y * 16;
  const int row = lane & 15;   // A row within tile; also B/D column within tile
  const int hi  = lane >> 4;

  // ---- TDM: stage W2 panel (rows n0..n0+15 of each gate) into LDS ----
  if (wave == 0) {
    // Tile: X = 512 halfs (contiguous), Y = 16 rows (stride HH), Z = 4 gates
    // (stride 512*HH). Element size 2 bytes. Tensor dims generous for OOB.
    const uint64_t ga = (uint64_t)(uintptr_t)(W2 + (size_t)n0 * HH); // byte addr
    const v4u g0 = { 1u,                                   // count=1, user desc
                     0u,                                   // lds_addr = 0
                     (unsigned int)ga,
                     (unsigned int)((ga >> 32) & 0x1FFFFFFu) | (2u << 30) }; // type=2
    const v8i g1 = { (int)(1u << 16),            // data_size = 1 (2 bytes)
                     (int)(512u << 16),          // tensor_dim0[15:0] = 512
                     (int)(2048u << 16),         // tensor_dim0 hi=0 | tensor_dim1 lo = 2048
                     (int)(512u << 16),          // tensor_dim1 hi=0 | tile_dim0 = 512
                     (int)(16u | (4u << 16)),    // tile_dim1 = 16 | tile_dim2 = 4
                     (int)512,                   // tensor_dim0_stride lo = 512
                     0,                          // stride0 hi=0 | stride1 lo16 = 0
                     4 };                        // tensor_dim1_stride >> 16 (262144>>16)
    const v4i g2 = { 4, 1, 0, 0 };               // tensor_dim2=4, tensor_dim3=1
    const v4i g3 = { 0, 0, 0, 0 };
    const v8i g4 = { 0, 0, 0, 0, 0, 0, 0, 0 };   // extra group (clang-23 6-arg form)
    __builtin_amdgcn_tensor_load_to_lds(g0, g1, g2, g3, g4, 0);
    __builtin_amdgcn_s_wait_tensorcnt(0);
  }
  __syncthreads();

  v8f acc0 = {}; // gate i
  v8f acc1 = {}; // gate f
  v8f acc2 = {}; // gate g
  v8f acc3 = {}; // gate o

  // ---- input contribution: A1 @ W1^T (global path) ----
#pragma unroll
  for (int k0 = 0; k0 < K1; k0 += 32) {
    v16h a  = load_a_tile(A1, lda1, m0, row, hi, k0);
    v16h b0 = load_b_tile(W1, K1, 0 * HH + n0 + row, hi, k0);
    v16h b1 = load_b_tile(W1, K1, 1 * HH + n0 + row, hi, k0);
    v16h b2 = load_b_tile(W1, K1, 2 * HH + n0 + row, hi, k0);
    v16h b3 = load_b_tile(W1, K1, 3 * HH + n0 + row, hi, k0);
    acc0 = __builtin_amdgcn_wmma_f32_16x16x32_f16(false, a, false, b0, (short)0, acc0, false, false);
    acc1 = __builtin_amdgcn_wmma_f32_16x16x32_f16(false, a, false, b1, (short)0, acc1, false, false);
    acc2 = __builtin_amdgcn_wmma_f32_16x16x32_f16(false, a, false, b2, (short)0, acc2, false, false);
    acc3 = __builtin_amdgcn_wmma_f32_16x16x32_f16(false, a, false, b3, (short)0, acc3, false, false);
  }

  // ---- recurrent contribution: h_prev @ W2^T (B operand from LDS) ----
#pragma unroll 4
  for (int k0 = 0; k0 < HH; k0 += 32) {
    v16h a  = load_a_tile(A2, HH, m0, row, hi, k0);
    const int kk = k0 + hi * 16;
    v16h b0 = *(const v16h*)(sB + (0 * 16 + row) * HH + kk);
    v16h b1 = *(const v16h*)(sB + (1 * 16 + row) * HH + kk);
    v16h b2 = *(const v16h*)(sB + (2 * 16 + row) * HH + kk);
    v16h b3 = *(const v16h*)(sB + (3 * 16 + row) * HH + kk);
    acc0 = __builtin_amdgcn_wmma_f32_16x16x32_f16(false, a, false, b0, (short)0, acc0, false, false);
    acc1 = __builtin_amdgcn_wmma_f32_16x16x32_f16(false, a, false, b1, (short)0, acc1, false, false);
    acc2 = __builtin_amdgcn_wmma_f32_16x16x32_f16(false, a, false, b2, (short)0, acc2, false, false);
    acc3 = __builtin_amdgcn_wmma_f32_16x16x32_f16(false, a, false, b3, (short)0, acc3, false, false);
  }

  // ---- pointwise LSTM cell update ----
  const int col = n0 + row;
  const float bi = bias[0 * HH + col];
  const float bf = bias[1 * HH + col];
  const float bg = bias[2 * HH + col];
  const float bo = bias[3 * HH + col];

#pragma unroll
  for (int v = 0; v < 8; ++v) {
    const int m = m0 + v + 8 * hi;
    const size_t idx = (size_t)m * HH + col;
    const float ig = sigm(acc0[v] + bi);
    const float fg = sigm(acc1[v] + bf);
    const float gg = tanhf(acc2[v] + bg);
    const float og = sigm(acc3[v] + bo);
    const float cc = fg * c[idx] + ig * gg;
    c[idx] = cc;
    hout[idx] = (_Float16)(og * tanhf(cc));
  }
}

// ---------------------------------------------------------------------------
// Utility kernels
// ---------------------------------------------------------------------------
__global__ void f32_to_f16_kernel(const float* __restrict__ s, _Float16* __restrict__ d, int n) {
  for (int i = blockIdx.x * blockDim.x + threadIdx.x; i < n; i += gridDim.x * blockDim.x)
    d[i] = (_Float16)s[i];
}

__global__ void combine_bias_kernel(const float* __restrict__ a, const float* __restrict__ b,
                                    float* __restrict__ o, int n) {
  int i = blockIdx.x * blockDim.x + threadIdx.x;
  if (i < n) o[i] = a[i] + b[i];
}

__global__ void zero_u32_kernel(unsigned* __restrict__ p, int n) {
  for (int i = blockIdx.x * blockDim.x + threadIdx.x; i < n; i += gridDim.x * blockDim.x)
    p[i] = 0u;
}

// out[b] = b_fc[0] + sum_k h[b,k] * w_fc[k]
__global__ void fc_kernel(const _Float16* __restrict__ h, const float* __restrict__ w,
                          const float* __restrict__ b, float* __restrict__ out) {
  int bidx = blockIdx.x * blockDim.x + threadIdx.x;
  if (bidx < BB) {
    float s = b[0];
    const _Float16* hr = h + (size_t)bidx * HH;
#pragma unroll 8
    for (int k = 0; k < HH; ++k) s += (float)hr[k] * w[k];
    out[bidx] = s;
  }
}

// ---------------------------------------------------------------------------
// Host launcher
// ---------------------------------------------------------------------------
static inline size_t align256(size_t x) { return (x + 255) & ~(size_t)255; }

extern "C" void kernel_launch(void* const* d_in, const int* in_sizes, int n_in,
                              void* d_out, int out_size, void* d_ws, size_t ws_size,
                              hipStream_t stream) {
  (void)in_sizes; (void)n_in; (void)out_size; (void)ws_size;

  const float* x      = (const float*)d_in[0];
  const float* w_ih_0 = (const float*)d_in[1];
  const float* w_hh_0 = (const float*)d_in[2];
  const float* b_ih_0 = (const float*)d_in[3];
  const float* b_hh_0 = (const float*)d_in[4];
  const float* w_ih_1 = (const float*)d_in[5];
  const float* w_hh_1 = (const float*)d_in[6];
  const float* b_ih_1 = (const float*)d_in[7];
  const float* b_hh_1 = (const float*)d_in[8];
  const float* w_fc   = (const float*)d_in[9];
  const float* b_fc   = (const float*)d_in[10];
  float* out = (float*)d_out;

  // ---- workspace carve-up ----
  char* ws = (char*)d_ws;
  size_t off = 0;
  auto carve = [&](size_t bytes) { char* p = ws + off; off += align256(bytes); return p; };

  _Float16* xh    = (_Float16*)carve((size_t)BB * TT * DD * 2);  // x as f16, layout [B,T,D]
  _Float16* wih0h = (_Float16*)carve((size_t)4 * HH * DD * 2);   // [2048, 64]
  _Float16* whh0h = (_Float16*)carve((size_t)4 * HH * HH * 2);   // [2048, 512]
  _Float16* wih1h = (_Float16*)carve((size_t)4 * HH * HH * 2);
  _Float16* whh1h = (_Float16*)carve((size_t)4 * HH * HH * 2);
  float*    bias0 = (float*)carve((size_t)4 * HH * 4);
  float*    bias1 = (float*)carve((size_t)4 * HH * 4);
  _Float16* h0[2];
  h0[0] = (_Float16*)carve((size_t)BB * HH * 2);
  h0[1] = (_Float16*)carve((size_t)BB * HH * 2);
  _Float16* h1[2];
  h1[0] = (_Float16*)carve((size_t)BB * HH * 2);
  h1[1] = (_Float16*)carve((size_t)BB * HH * 2);
  float* c0 = (float*)carve((size_t)BB * HH * 4);
  float* c1 = (float*)carve((size_t)BB * HH * 4);

  // ---- one-time conversions (every call: launcher must be stateless) ----
  auto cvt = [&](const float* s, _Float16* d, int n) {
    int blocks = (n + 255) / 256; if (blocks > 4096) blocks = 4096;
    f32_to_f16_kernel<<<blocks, 256, 0, stream>>>(s, d, n);
  };
  cvt(x,      xh,    BB * TT * DD);
  cvt(w_ih_0, wih0h, 4 * HH * DD);
  cvt(w_hh_0, whh0h, 4 * HH * HH);
  cvt(w_ih_1, wih1h, 4 * HH * HH);
  cvt(w_hh_1, whh1h, 4 * HH * HH);
  combine_bias_kernel<<<(4 * HH + 255) / 256, 256, 0, stream>>>(b_ih_0, b_hh_0, bias0, 4 * HH);
  combine_bias_kernel<<<(4 * HH + 255) / 256, 256, 0, stream>>>(b_ih_1, b_hh_1, bias1, 4 * HH);

  // ---- zero initial state (h0[0], h1[0], c0, c1) ----
  auto zero = [&](void* p, size_t bytes) {
    int n = (int)(bytes / 4);
    int blocks = (n + 255) / 256; if (blocks > 2048) blocks = 2048;
    zero_u32_kernel<<<blocks, 256, 0, stream>>>((unsigned*)p, n);
  };
  zero(h0[0], (size_t)BB * HH * 2);
  zero(h1[0], (size_t)BB * HH * 2);
  zero(c0,    (size_t)BB * HH * 4);
  zero(c1,    (size_t)BB * HH * 4);

  // ---- sequential recurrence ----
  dim3 grid(4, 32), block(128);
  for (int t = 0; t < TT; ++t) {
    const int in = t & 1, nx = (t + 1) & 1;
    // layer 0: A1 = x[:, t, :] (f16, row stride T*D), A2 = h0 prev
    lstm_step_kernel<DD><<<grid, block, 0, stream>>>(
        xh + (size_t)t * DD, TT * DD, wih0h,
        h0[in], whh0h, bias0, c0, h0[nx]);
    // layer 1: A1 = h0 new (row stride H), A2 = h1 prev
    lstm_step_kernel<HH><<<grid, block, 0, stream>>>(
        h0[nx], HH, wih1h,
        h1[in], whh1h, bias1, c1, h1[nx]);
  }

  // after t = T-1 (odd), final h1 lives in buffer (T)&1 == 0
  fc_kernel<<<1, BB, 0, stream>>>(h1[TT & 1], w_fc, b_fc, out);
}
